// NTMLayer_51908974739761
// MI455X (gfx1250) — matrix operand order repaired and checked by reading, not compiled, not run on hardware
//
#include <hip/hip_runtime.h>
#include <hip/hip_bf16.h>
#include <math.h>

// Problem dims (from reference): B=8192, D=256, L=128, C=64, H=100
#define NB   8192
#define ND   256
#define NL   128
#define NC   64
#define NH_  100
#define KF   320   // D + C (feats width)
#define HP   112   // H padded to 7 WMMA N-tiles
#define KH   128   // padded K for layer2 / heads (4 WMMA K-steps)
#define NHD  256   // 4 heads x 64 columns
#define ROWS 64    // batch rows per workgroup in GEMM kernel
#define SAK  328   // padded K-stride for K=320 panels (16B-aligned rows)
#define SHK  136   // padded K-stride for K=128 panels (16B-aligned rows)
#define EPSF 1e-12f

typedef __attribute__((ext_vector_type(16))) __bf16 v16bf;
typedef __attribute__((ext_vector_type(8)))  __bf16 v8bf;
typedef __attribute__((ext_vector_type(8)))  float  v8f;
typedef __attribute__((ext_vector_type(4)))  unsigned int u32x4;
typedef __attribute__((ext_vector_type(8)))  unsigned int u32x8;

// ---------------- WMMA fragment loaders (CDNA5 16x16x32 bf16 layouts) -------
// A (MxK row-major in LDS): lane = h*16+m needs K = {h*8..h*8+7} and
// {16+h*8..16+h*8+7} -> two contiguous 16B runs -> 2x ds_load_b128.
__device__ inline v16bf load_a_frag(const __bf16* base, int stride, int row0,
                                    int k0, int lane) {
  int m = lane & 15, h = lane >> 4;
  const __bf16* p = base + (row0 + m) * stride + k0 + h * 8;
  v8bf lo = *(const v8bf*)p;         // K = h*8 .. h*8+7
  v8bf hi = *(const v8bf*)(p + 16);  // K = 16+h*8 .. 16+h*8+7
  return __builtin_shufflevector(lo, hi, 0, 1, 2, 3, 4, 5, 6, 7, 8, 9, 10, 11,
                                 12, 13, 14, 15);
}

// B staged TRANSPOSED in LDS as [N][K]: lane = h*16+n needs K = {h*8..h*8+7}
// and {16+h*8..16+h*8+7} -> two contiguous 16B runs -> 2x ds_load_b128.
__device__ inline v16bf load_bt_frag(const __bf16* base, int stride, int k0,
                                     int n0, int lane) {
  int n = lane & 15, h = lane >> 4;
  const __bf16* p = base + (n0 + n) * stride + k0 + h * 8;
  v8bf lo = *(const v8bf*)p;
  v8bf hi = *(const v8bf*)(p + 16);
  return __builtin_shufflevector(lo, hi, 0, 1, 2, 3, 4, 5, 6, 7, 8, 9, 10, 11,
                                 12, 13, 14, 15);
}

__device__ inline float sigmoidf(float x) { return 1.f / (1.f + expf(-x)); }

// ======================= Kernel 1: controller + heads =======================
__global__ __launch_bounds__(128) void ntm_controller_kernel(
    const float* __restrict__ inputs, const float* __restrict__ reading,
    const float* __restrict__ W1, const float* __restrict__ b1,
    const float* __restrict__ W2, const float* __restrict__ b2,
    const float* __restrict__ Wk_r, const float* __restrict__ bk_r,
    const float* __restrict__ Ws_r, const float* __restrict__ bs_r,
    const float* __restrict__ Wk_w, const float* __restrict__ bk_w,
    const float* __restrict__ Ws_w, const float* __restrict__ bs_w,
    const float* __restrict__ We_w, const float* __restrict__ be_w,
    const float* __restrict__ Wa_w, const float* __restrict__ ba_w,
    float* __restrict__ kr, float* __restrict__ kw,
    float* __restrict__ ev, float* __restrict__ av,
    float* __restrict__ sr, float* __restrict__ sw) {
  __shared__ __bf16 sA[ROWS * SAK];  // feats tile, padded stride     41 KB
  __shared__ __bf16 sW[HP * SAK];    // transposed weight panels      72 KB
  __shared__ __bf16 sH[ROWS * SHK];  // h1 (zero-padded K)            17 KB
  __shared__ __bf16 sC[ROWS * SHK];  // ctr (zero-padded K)           17 KB

  const int tid  = threadIdx.x;
  const int wid  = tid >> 5;
  const int lane = tid & 31;
  const int row0 = blockIdx.x * ROWS;
  const int wrow = wid * 16;
  const int nl   = lane & 15;
  const int hh   = lane >> 4;

  // Stage feats = concat(inputs, reading) as bf16
  for (int i = tid; i < ROWS * KF; i += 128) {
    int r = i / KF, c = i % KF;
    float v = (c < ND) ? inputs[(size_t)(row0 + r) * ND + c]
                       : reading[(size_t)(row0 + r) * NC + (c - ND)];
    sA[r * SAK + c] = (__bf16)v;
  }
  // Stage W1^T as [112][320] (rows >= 100 zero)
  for (int i = tid; i < HP * KF; i += 128) {
    int n = i / KF, k = i % KF;
    sW[n * SAK + k] = (__bf16)((n < NH_) ? W1[k * NH_ + n] : 0.f);
  }
  // Zero-fill padded activation buffers
  for (int i = tid; i < ROWS * SHK; i += 128) {
    sH[i] = (__bf16)0.f;
    sC[i] = (__bf16)0.f;
  }
  __syncthreads();

  // ---- layer 1: h1 = tanh(feats @ W1 + b1), K = 320 ----
  for (int nt = 0; nt < 7; ++nt) {
    v8f acc = {};
    for (int kt = 0; kt < KF; kt += 32) {
      v16bf af = load_a_frag(sA, SAK, wrow, kt, lane);
      v16bf bf = load_bt_frag(sW, SAK, kt, nt * 16, lane);
      acc = __builtin_amdgcn_wmma_f32_16x16x32_bf16(false, af, false, bf,
                                                    (short)0, acc, false, false);
    }
    int col = nt * 16 + nl;
    float bias = (col < NH_) ? b1[col] : 0.f;
#pragma unroll
    for (int r = 0; r < 8; ++r) {
      int rr = wrow + r + 8 * hh;
      sH[rr * SHK + col] = (__bf16)tanhf(acc[r] + bias);
    }
  }
  __syncthreads();

  // Stage W2^T as [112][128] zero-padded
  for (int i = tid; i < HP * KH; i += 128) {
    int n = i / KH, k = i % KH;
    sW[n * SHK + k] = (__bf16)((n < NH_ && k < NH_) ? W2[k * NH_ + n] : 0.f);
  }
  __syncthreads();

  // ---- layer 2: ctr = tanh(h1 @ W2 + b2), K = 128 (padded) ----
  for (int nt = 0; nt < 7; ++nt) {
    v8f acc = {};
    for (int kt = 0; kt < KH; kt += 32) {
      v16bf af = load_a_frag(sH, SHK, wrow, kt, lane);
      v16bf bf = load_bt_frag(sW, SHK, kt, nt * 16, lane);
      acc = __builtin_amdgcn_wmma_f32_16x16x32_bf16(false, af, false, bf,
                                                    (short)0, acc, false, false);
    }
    int col = nt * 16 + nl;
    float bias = (col < NH_) ? b2[col] : 0.f;
#pragma unroll
    for (int r = 0; r < 8; ++r) {
      int rr = wrow + r + 8 * hh;
      sC[rr * SHK + col] = (__bf16)tanhf(acc[r] + bias);
    }
  }
  __syncthreads();

  // Stage fused head weights^T [256][128] = rows of [Wk_r|Wk_w|We_w|Wa_w]
  for (int i = tid; i < NHD * KH; i += 128) {
    int n = i / KH, k = i % KH;
    int head = n >> 6, c = n & 63;
    const float* Wm = (head == 0) ? Wk_r : (head == 1) ? Wk_w
                     : (head == 2) ? We_w : Wa_w;
    sW[n * SHK + k] = (__bf16)((k < NH_) ? Wm[k * NC + c] : 0.f);
  }
  __syncthreads();

  // ---- heads: ctr @ Wh, K = 128, N = 256 ----
  for (int nt = 0; nt < 16; ++nt) {
    v8f acc = {};
    for (int kt = 0; kt < KH; kt += 32) {
      v16bf af = load_a_frag(sC, SHK, wrow, kt, lane);
      v16bf bf = load_bt_frag(sW, SHK, kt, nt * 16, lane);
      acc = __builtin_amdgcn_wmma_f32_16x16x32_bf16(false, af, false, bf,
                                                    (short)0, acc, false, false);
    }
    int col = nt * 16 + nl;
    int head = col >> 6, c = col & 63;
    float bias = ((head == 0) ? bk_r : (head == 1) ? bk_w
                 : (head == 2) ? be_w : ba_w)[c];
    float* dst = (head == 0) ? kr : (head == 1) ? kw : (head == 2) ? ev : av;
#pragma unroll
    for (int r = 0; r < 8; ++r) {
      int rr = wrow + r + 8 * hh;
      float v = acc[r] + bias;
      if (head >= 2) v = sigmoidf(v);  // eraser / adder get sigmoid
      dst[(size_t)(row0 + rr) * NC + c] = v;
    }
  }

  // ---- strengths: s = sigmoid(ctr . Ws + bs) ----
  if (tid < 2 * ROWS) {
    int r = tid & 63;
    bool isw = tid >= ROWS;
    const float* Wsv = isw ? Ws_w : Ws_r;
    float acc = isw ? bs_w[0] : bs_r[0];
    for (int k = 0; k < NH_; ++k) acc += (float)sC[r * SHK + k] * Wsv[k];
    (isw ? sw : sr)[row0 + r] = sigmoidf(acc);
  }
}

// ====================== Kernel 2: fused NTM memory update ===================
__device__ inline float block_reduce_sum(float v, float* sred, int tid) {
  sred[tid] = v;
  __syncthreads();
#pragma unroll
  for (int s = 128; s > 0; s >>= 1) {
    if (tid < s) sred[tid] += sred[tid + s];
    __syncthreads();
  }
  float r = sred[0];
  __syncthreads();
  return r;
}

__device__ inline float block_reduce_max(float v, float* sred, int tid) {
  sred[tid] = v;
  __syncthreads();
#pragma unroll
  for (int s = 128; s > 0; s >>= 1) {
    if (tid < s) sred[tid] = fmaxf(sred[tid], sred[tid + s]);
    __syncthreads();
  }
  float r = sred[0];
  __syncthreads();
  return r;
}

// Dynamic-LDS layout (no static __shared__ -> dynamic region at LDS offset 0,
// which is what the TDM descriptor's lds_addr field refers to).
#define OFF_SMEM 0
#define OFF_SRED (NL * NC * 4)            // 32768
#define OFF_SKR  (OFF_SRED + 256 * 4)     // +1KB
#define OFF_SKW  (OFF_SKR + NC * 4)
#define OFF_SE   (OFF_SKW + NC * 4)
#define OFF_SAV  (OFF_SE + NC * 4)
#define OFF_SLR  (OFF_SAV + NC * 4)
#define OFF_SLW  (OFF_SLR + NL * 4)
#define OFF_SWR  (OFF_SLW + NL * 4)
#define OFF_SWW  (OFF_SWR + NL * 4)
#define LDS2_SZ  (OFF_SWW + NL * 4)       // 36864 bytes

__global__ __launch_bounds__(256) void ntm_memory_kernel(
    const float* __restrict__ memory,
    const float* __restrict__ kr, const float* __restrict__ kw,
    const float* __restrict__ ev, const float* __restrict__ av,
    const float* __restrict__ sr, const float* __restrict__ sw,
    float* __restrict__ out_read, float* __restrict__ out_mem) {
  extern __shared__ char lds[];
  float* smem = (float*)(lds + OFF_SMEM);  // 128x64 memory tile (32 KB)
  float* sred = (float*)(lds + OFF_SRED);
  float* skr  = (float*)(lds + OFF_SKR);
  float* skw  = (float*)(lds + OFF_SKW);
  float* se   = (float*)(lds + OFF_SE);
  float* sa   = (float*)(lds + OFF_SAV);
  float* sLr  = (float*)(lds + OFF_SLR);
  float* sLw  = (float*)(lds + OFF_SLW);
  float* sWr  = (float*)(lds + OFF_SWR);
  float* sWw  = (float*)(lds + OFF_SWW);

  const int b = blockIdx.x, tid = threadIdx.x;
  float4* sm4 = (float4*)smem;

  // ---- TDM: DMA memory[b] (8192 x fp32) -> LDS offset 0, one descriptor ----
  if (tid == 0) {
    unsigned long long ga =
        (unsigned long long)(const void*)(memory + (size_t)b * (NL * NC));
    u32x4 g0;
    g0[0] = 1u;                       // count=1, user descriptor
    g0[1] = 0u;                       // lds_addr = 0 (dynamic LDS base)
    g0[2] = (unsigned int)ga;         // global_addr[31:0]
    g0[3] = ((unsigned int)(ga >> 32) & 0x01FFFFFFu) | (2u << 30);  // type=2
    u32x8 g1;
    g1[0] = (2u << 16);               // data_size = 2 (4 bytes/elem)
    g1[1] = (8192u & 0xFFFFu) << 16;  // tensor_dim0[15:0]
    g1[2] = (8192u >> 16) | (1u << 16);         // dim0 hi | tensor_dim1 = 1
    g1[3] = (8192u & 0xFFFFu) << 16;            // tile_dim0 = 8192
    g1[4] = 1u;                       // tile_dim1 = 1, tile_dim2 = 0
    g1[5] = 8192u;                    // tensor_dim0_stride lo
    g1[6] = 0u;
    g1[7] = 0u;
    u32x4 gz = {0u, 0u, 0u, 0u};
    asm volatile("tensor_load_to_lds %0, %1, %2, %3"
                 :
                 : "s"(g0), "s"(g1), "s"(gz), "s"(gz)
                 : "memory");
    asm volatile("s_wait_tensorcnt 0x0" ::: "memory");
  }
  if (tid < NC) {  // head vectors overlap the DMA
    size_t o = (size_t)b * NC + tid;
    skr[tid] = kr[o];
    skw[tid] = kw[o];
    se[tid]  = ev[o];
    sa[tid]  = av[o];
  }
  __syncthreads();

  // Frobenius sum-of-squares straight out of LDS (b128 reads)
  float ss = 0.f;
#pragma unroll
  for (int j = 0; j < 8; ++j) {
    float4 q = sm4[tid + 256 * j];
    ss += q.x * q.x + q.y * q.y + q.z * q.z + q.w * q.w;
  }
  float mss = block_reduce_sum(ss, sred, tid);
  float inv_m = 1.f / fmaxf(sqrtf(mss), EPSF);

  float kr2 = block_reduce_sum((tid < NC) ? skr[tid] * skr[tid] : 0.f, sred, tid);
  float kw2 = block_reduce_sum((tid < NC) ? skw[tid] * skw[tid] : 0.f, sred, tid);
  float inv_kr = 1.f / fmaxf(sqrtf(kr2), EPSF);
  float inv_kw = 1.f / fmaxf(sqrtf(kw2), EPSF);
  float s_rv = sr[b], s_wv = sw[b];

  // Cosine-similarity logits per memory line (both heads, one LDS pass)
  if (tid < NL) {
    const float* mrow = &smem[tid * NC];
    float dr = 0.f, dw = 0.f;
#pragma unroll 8
    for (int c = 0; c < NC; ++c) {
      float m = mrow[c];
      dr += m * skr[c];
      dw += m * skw[c];
    }
    sLr[tid] = dr * inv_m * inv_kr * s_rv;
    sLw[tid] = dw * inv_m * inv_kw * s_wv;
  }
  __syncthreads();

  // Softmax over L for both heads
  float mr = block_reduce_max((tid < NL) ? sLr[tid] : -3.0e38f, sred, tid);
  float er_ = (tid < NL) ? expf(sLr[tid] - mr) : 0.f;
  float sumr = block_reduce_sum(er_, sred, tid);
  if (tid < NL) sWr[tid] = er_ / sumr;

  float mw = block_reduce_max((tid < NL) ? sLw[tid] : -3.0e38f, sred, tid);
  float ew_ = (tid < NL) ? expf(sLw[tid] - mw) : 0.f;
  float sumw = block_reduce_sum(ew_, sred, tid);
  if (tid < NL) sWw[tid] = ew_ / sumw;
  __syncthreads();

  // Weighted read: read[b,c] = sum_l mem[l,c] * w_r[l]
  if (tid < NC) {
    float acc = 0.f;
#pragma unroll 8
    for (int l = 0; l < NL; ++l) acc += smem[l * NC + tid] * sWr[l];
    out_read[(size_t)b * NC + tid] = acc;
  }

  // Erase/add update streamed out with b128 stores
  float4* dst = (float4*)(out_mem + (size_t)b * (NL * NC));
#pragma unroll
  for (int j = 0; j < 8; ++j) {
    int v = tid + 256 * j;
    int idx0 = v * 4;
    int l = idx0 >> 6, c0 = idx0 & 63;
    float wwv = sWw[l];
    float4 q = sm4[v];
    q.x = q.x * (1.f - wwv * se[c0 + 0]) + wwv * sa[c0 + 0];
    q.y = q.y * (1.f - wwv * se[c0 + 1]) + wwv * sa[c0 + 1];
    q.z = q.z * (1.f - wwv * se[c0 + 2]) + wwv * sa[c0 + 2];
    q.w = q.w * (1.f - wwv * se[c0 + 3]) + wwv * sa[c0 + 3];
    dst[v] = q;
  }
}

// ============================== launcher ====================================
extern "C" void kernel_launch(void* const* d_in, const int* in_sizes, int n_in,
                              void* d_out, int out_size, void* d_ws,
                              size_t ws_size, hipStream_t stream) {
  const float* inputs  = (const float*)d_in[0];
  const float* memory  = (const float*)d_in[1];
  const float* reading = (const float*)d_in[2];
  const float* W1   = (const float*)d_in[3];
  const float* b1   = (const float*)d_in[4];
  const float* W2   = (const float*)d_in[5];
  const float* b2   = (const float*)d_in[6];
  const float* Wk_r = (const float*)d_in[7];
  const float* bk_r = (const float*)d_in[8];
  const float* Ws_r = (const float*)d_in[9];
  const float* bs_r = (const float*)d_in[10];
  const float* Wk_w = (const float*)d_in[11];
  const float* bk_w = (const float*)d_in[12];
  const float* Ws_w = (const float*)d_in[13];
  const float* bs_w = (const float*)d_in[14];
  const float* We_w = (const float*)d_in[15];
  const float* be_w = (const float*)d_in[16];
  const float* Wa_w = (const float*)d_in[17];
  const float* ba_w = (const float*)d_in[18];

  float* out_read = (float*)d_out;
  float* out_mem  = out_read + (size_t)NB * NC;

  float* ws = (float*)d_ws;
  const size_t BC = (size_t)NB * NC;
  float* kr = ws;
  float* kw = ws + BC;
  float* ev = ws + 2 * BC;
  float* av = ws + 3 * BC;
  float* sr = ws + 4 * BC;
  float* sw = sr + NB;

  ntm_controller_kernel<<<NB / ROWS, 128, 0, stream>>>(
      inputs, reading, W1, b1, W2, b2, Wk_r, bk_r, Ws_r, bs_r, Wk_w, bk_w,
      Ws_w, bs_w, We_w, be_w, Wa_w, ba_w, kr, kw, ev, av, sr, sw);

  ntm_memory_kernel<<<NB, 256, LDS2_SZ, stream>>>(memory, kr, kw, ev, av, sr,
                                                  sw, out_read, out_mem);
}
